// IndexedMatMul_56676388438552
// MI455X (gfx1250) — matrix operand order repaired and checked by reading, not compiled, not run on hardware
//
#include <hip/hip_runtime.h>

// Problem (from reference): X[4,2048,1024] f32, W[8,1024,1024] f32,
// indices[4,2048] i32.  Y[t,:] = X[t,:] @ W[idx[t]]^T
#define TOKENS    8192
#define DIM       1024
#define N_EXPERTS 8
#define RPB       64                 // rows (tokens) per block: 4 x 16-row WMMA tiles
#define MAX_RBLK  136                // <= 8192/64 + 8 padded segments
#define PERM_CAP  (MAX_RBLK * RPB)

// Workspace layout (ints)
#define WS_COUNTS 0                  // [8]  per-expert token counts
#define WS_CURSOR 8                  // [8]  scatter cursors
#define WS_POFF   16                 // [8]  per-expert first row-block
#define WS_NRBLK  24                 // [1]  total row blocks
#define WS_TEXP   32                 // [MAX_RBLK] row-block -> expert
#define WS_PERM   (32 + MAX_RBLK)    // [PERM_CAP] row -> token id (-1 = pad)

typedef __attribute__((ext_vector_type(16))) _Float16 v16h;
typedef __attribute__((ext_vector_type(8)))  float    v8f;

__global__ void moe_init(int* ws) {
  int i = blockIdx.x * blockDim.x + threadIdx.x;
  if (i < PERM_CAP) ws[WS_PERM + i] = -1;
  if (i < 16)       ws[WS_COUNTS + i] = 0;   // counts + cursors
  if (i == 16)      ws[WS_NRBLK] = 0;
}

__global__ void moe_hist(const int* __restrict__ idx, int* ws) {
  int t = blockIdx.x * blockDim.x + threadIdx.x;
  if (t < TOKENS) atomicAdd(&ws[WS_COUNTS + idx[t]], 1);
}

__global__ void moe_scan(int* ws) {
  if (blockIdx.x == 0 && threadIdx.x == 0) {
    int off = 0;
    for (int e = 0; e < N_EXPERTS; ++e) {
      ws[WS_POFF + e] = off;
      int nb = (ws[WS_COUNTS + e] + RPB - 1) / RPB;
      for (int b = 0; b < nb; ++b) ws[WS_TEXP + off + b] = e;
      off += nb;
    }
    ws[WS_NRBLK] = off;
  }
}

__global__ void moe_scatter(const int* __restrict__ idx, int* ws) {
  int t = blockIdx.x * blockDim.x + threadIdx.x;
  if (t < TOKENS) {
    int e   = idx[t];
    int pos = atomicAdd(&ws[WS_CURSOR + e], 1);
    ws[WS_PERM + ws[WS_POFF + e] * RPB + pos] = t;
  }
}

// GEMM: block = 256 threads (8 wave32), covers 64 rows x 128 cols.
// Wave w: row-tile (w&3), col-half (w>>2); 4 f32 accumulators (16x64 slab).
// K loop: 32 chunks of K=32, double-buffered via GLOBAL_LOAD_ASYNC_TO_LDS_B128.
// LDS chunk layout (f32, stride 36 dwords for conflict-free ds_load_b128):
//   sX[buf]: 64 rows  x 32 k  @ smem + buf*2304
//   sW[buf]: 128 cols x 32 k  @ smem + 4608 + buf*4608
#define XCH 2304                     // 64*36 floats
#define WCH 4608                     // 128*36 floats
#define SMEM_FLTS (2*XCH + 2*WCH)    // 13824 floats = 55296 B

__global__ __launch_bounds__(256)
void moe_gemm(const float* __restrict__ X, const float* __restrict__ W,
              const int* __restrict__ ws, float* __restrict__ Y) {
  __shared__ float smem[SMEM_FLTS];
  __shared__ int   sRow[RPB];

  const int rb = blockIdx.x;
  if (rb >= ws[WS_NRBLK]) return;                    // uniform exit
  const int e = ws[WS_TEXP + rb];

  if (threadIdx.x < RPB) sRow[threadIdx.x] = ws[WS_PERM + rb * RPB + threadIdx.x];
  __syncthreads();

  const unsigned ldsBase = (unsigned)(unsigned long long)(void*)&smem[0];
  const unsigned long long Xp = (unsigned long long)X;
  const unsigned long long Wp = (unsigned long long)(W + (size_t)e * DIM * DIM);
  const unsigned colBase = blockIdx.y * 128u;

  // Issue one K-chunk (6 async b128 per thread: 2 for X rows, 4 for W cols).
  auto issue_chunk = [&](int kc, int buf) {
    unsigned ldsX = ldsBase + (unsigned)buf * (XCH * 4u);
#pragma unroll
    for (int t = 0; t < 2; ++t) {
      int q = threadIdx.x + t * 256;
      int r = q >> 3, c4 = q & 7;                    // 8 x 16B per row
      int tok = sRow[r]; if (tok < 0) tok = 0;       // pad rows: load row 0, never stored
      unsigned goff = (unsigned)tok * (DIM * 4u) + (unsigned)(kc + c4 * 4) * 4u;
      unsigned loff = ldsX + (unsigned)(r * 36 + c4 * 4) * 4u;
      asm volatile("global_load_async_to_lds_b128 %0, %1, %2 offset:0"
                   :: "v"(loff), "v"(goff), "s"(Xp) : "memory");
    }
    unsigned ldsW = ldsBase + (2u * XCH + (unsigned)buf * WCH) * 4u;
#pragma unroll
    for (int t = 0; t < 4; ++t) {
      int q = threadIdx.x + t * 256;
      int c = q >> 3, c4 = q & 7;
      unsigned goff = (colBase + (unsigned)c) * (DIM * 4u) + (unsigned)(kc + c4 * 4) * 4u;
      unsigned loff = ldsW + (unsigned)(c * 36 + c4 * 4) * 4u;
      asm volatile("global_load_async_to_lds_b128 %0, %1, %2 offset:0"
                   :: "v"(loff), "v"(goff), "s"(Wp) : "memory");
    }
  };

  const int lane = threadIdx.x & 31;
  const int l16  = lane & 15;
  const int half = lane >> 4;                        // 0: lanes 0-15, 1: lanes 16-31
  const int wave = threadIdx.x >> 5;
  const int rt   = wave & 3;                         // row tile 0..3
  const int ch   = wave >> 2;                        // col half 0..1

  v8f acc[4] = {v8f{}, v8f{}, v8f{}, v8f{}};

  issue_chunk(0, 0);
  for (int c = 0; c < DIM / 32; ++c) {
    const int buf = c & 1;
    if (c + 1 < DIM / 32) {
      issue_chunk((c + 1) * 32, (c + 1) & 1);
      asm volatile("s_wait_asynccnt 6" ::: "memory");  // my chunk-c copies done
    } else {
      asm volatile("s_wait_asynccnt 0" ::: "memory");
    }
    __syncthreads();                                   // everyone's chunk c visible

    const float* sXb = &smem[buf * XCH];
    const float* sWb = &smem[2 * XCH + buf * WCH];

    // A fragment (16-bit A 16x32 layout): lane<16 -> K 0-7 & 16-23,
    // lane>=16 -> K 8-15 & 24-31, row = rt*16 + (lane&15).  f32->f16 here.
    const float* pA = sXb + (rt * 16 + l16) * 36 + half * 8;
    float4 a0 = *(const float4*)(pA);
    float4 a1 = *(const float4*)(pA + 4);
    float4 a2 = *(const float4*)(pA + 16);
    float4 a3 = *(const float4*)(pA + 20);
    v16h a = { (_Float16)a0.x, (_Float16)a0.y, (_Float16)a0.z, (_Float16)a0.w,
               (_Float16)a1.x, (_Float16)a1.y, (_Float16)a1.z, (_Float16)a1.w,
               (_Float16)a2.x, (_Float16)a2.y, (_Float16)a2.z, (_Float16)a2.w,
               (_Float16)a3.x, (_Float16)a3.y, (_Float16)a3.z, (_Float16)a3.w };

#pragma unroll
    for (int nt = 0; nt < 4; ++nt) {
      // B fragment (16-bit B 32x16): lane<16 -> K 0-7 & 16-23, lane>=16 -> K 8-15 & 24-31 of column
      // n = ch*64 + nt*16 + (lane&15); B column == W row (Y = X W^T).
      const float* pB = sWb + (ch * 64 + nt * 16 + l16) * 36 + half * 8;
      float4 b0 = *(const float4*)(pB);
      float4 b1 = *(const float4*)(pB + 4);
      float4 b2 = *(const float4*)(pB + 16);
      float4 b3 = *(const float4*)(pB + 20);
      v16h b = { (_Float16)b0.x, (_Float16)b0.y, (_Float16)b0.z, (_Float16)b0.w,
                 (_Float16)b1.x, (_Float16)b1.y, (_Float16)b1.z, (_Float16)b1.w,
                 (_Float16)b2.x, (_Float16)b2.y, (_Float16)b2.z, (_Float16)b2.w,
                 (_Float16)b3.x, (_Float16)b3.y, (_Float16)b3.z, (_Float16)b3.w };
      acc[nt] = __builtin_amdgcn_wmma_f32_16x16x32_f16(
                  false, a, false, b, (short)0, acc[nt], false, false);
    }
    __syncthreads();                                   // buf reusable for chunk c+2
  }

  // C/D layout: VGPR j -> M = j + 8*half, N = lane&15 per 16-col subtile.
#pragma unroll
  for (int nt = 0; nt < 4; ++nt) {
    int col = (int)colBase + ch * 64 + nt * 16 + l16;
#pragma unroll
    for (int j = 0; j < 8; ++j) {
      int tok = sRow[rt * 16 + half * 8 + j];
      if (tok >= 0) Y[(size_t)tok * DIM + col] = acc[nt][j];
    }
  }
}

extern "C" void kernel_launch(void* const* d_in, const int* in_sizes, int n_in,
                              void* d_out, int out_size, void* d_ws, size_t ws_size,
                              hipStream_t stream) {
  const float* X   = (const float*)d_in[0];
  const float* W   = (const float*)d_in[1];
  const int*   idx = (const int*)d_in[2];
  float*       Y   = (float*)d_out;
  int*         ws  = (int*)d_ws;

  moe_init<<<(PERM_CAP + 255) / 256, 256, 0, stream>>>(ws);
  moe_hist<<<TOKENS / 256, 256, 0, stream>>>(idx, ws);
  moe_scan<<<1, 32, 0, stream>>>(ws);
  moe_scatter<<<TOKENS / 256, 256, 0, stream>>>(idx, ws);

  dim3 grid(MAX_RBLK, DIM / 128, 1);
  moe_gemm<<<grid, 256, 0, stream>>>(X, W, ws, Y);
}
